// PhimoeDecoderLayer_6975026888749
// MI455X (gfx1250) — matrix operand order repaired and checked by reading, not compiled, not run on hardware
//
#include <hip/hip_runtime.h>
#include <math.h>

typedef __attribute__((ext_vector_type(16))) _Float16 v16h;
typedef __attribute__((ext_vector_type(8)))  float    v8f;

#define S_LEN   1024
#define HDIM    2048
#define KVD     512
#define NHEADS  32
#define KVHEADS 8
#define DHEAD   64
#define NEXP    8
#define IDIM    4096
#define EPS_LN  1e-5f
#define JITTER  0.01f

__device__ __forceinline__ float warp_reduce_sum32(float v) {
  #pragma unroll
  for (int off = 16; off > 0; off >>= 1) v += __shfl_xor(v, off, 32);
  return v;
}

// ---------------- LayerNorm ----------------
__global__ __launch_bounds__(256) void ln_kernel(
    const float* __restrict__ x, const float* __restrict__ w,
    const float* __restrict__ b, float* __restrict__ y) {
  int t = blockIdx.x;
  const float* xr = x + (size_t)t * HDIM;
  float s = 0.f, sq = 0.f;
  for (int i = threadIdx.x; i < HDIM; i += blockDim.x) { float v = xr[i]; s += v; sq += v * v; }
  __shared__ float rs[16], rq[16];
  s  = warp_reduce_sum32(s);
  sq = warp_reduce_sum32(sq);
  int wid = threadIdx.x >> 5, lane = threadIdx.x & 31;
  if (lane == 0) { rs[wid] = s; rq[wid] = sq; }
  __syncthreads();
  if (threadIdx.x == 0) {
    float a = 0.f, c = 0.f;
    int nw = blockDim.x >> 5;
    for (int i = 0; i < nw; ++i) { a += rs[i]; c += rq[i]; }
    rs[0] = a; rq[0] = c;
  }
  __syncthreads();
  float mu  = rs[0] * (1.f / HDIM);
  float var = rq[0] * (1.f / HDIM) - mu * mu;
  float inv = rsqrtf(fmaxf(var, 0.f) + EPS_LN);
  float* yr = y + (size_t)t * HDIM;
  for (int i = threadIdx.x; i < HDIM; i += blockDim.x)
    yr[i] = (xr[i] - mu) * inv * w[i] + b[i];
}

// ---------------- Generic WMMA GEMM:  C[M,N] = A[M,K] @ W[N,K]^T ----------------
// Per wave: 16x64 output tile (4 N-subtiles sharing one A fragment -> 4 WMMAs/k-step).
// Block (8 waves): 16 x 512. Requires N % 512 == 0, K % 32 == 0, M % 16 == 0.
// Epilogue: (+bias[n]) (*rowscale[m*rs_stride]) (+residual[m,n]) (+=C if accumulate)
__global__ __launch_bounds__(256) void gemm_wmma_f16(
    const float* __restrict__ A, const float* __restrict__ W,
    const float* __restrict__ bias,
    const float* __restrict__ residual,
    const float* __restrict__ rowscale, int rs_stride,
    float* __restrict__ C,
    int M, int N, int K, int accumulate) {
  int lane = threadIdx.x & 31;
  int wv   = threadIdx.x >> 5;
  int half = lane >> 4;
  int mr   = lane & 15;
  int n0 = blockIdx.x * 512 + wv * 64;
  int m0 = blockIdx.y * 16;
  if (n0 >= N || m0 >= M) return;     // wave-uniform: EXEC stays all-ones per wave

  v8f acc0 = {}, acc1 = {}, acc2 = {}, acc3 = {};
  const float* Arow = A + (size_t)(m0 + mr) * K;
  const float* Wr0  = W + (size_t)(n0 + mr) * K;
  const float* Wr1  = Wr0 + (size_t)16 * K;
  const float* Wr2  = Wr0 + (size_t)32 * K;
  const float* Wr3  = Wr0 + (size_t)48 * K;
  for (int kb = 0; kb < K; kb += 32) {
    v16h a;
    #pragma unroll
    for (int j = 0; j < 8; ++j) {
      a[j]     = (_Float16)Arow[kb + half * 16 + j];
      a[j + 8] = (_Float16)Arow[kb + half * 16 + 8 + j];
    }
    v16h b0, b1, b2, b3;
    #pragma unroll
    for (int j = 0; j < 16; ++j) {
      b0[j] = (_Float16)Wr0[kb + half * 16 + j];
      b1[j] = (_Float16)Wr1[kb + half * 16 + j];
      b2[j] = (_Float16)Wr2[kb + half * 16 + j];
      b3[j] = (_Float16)Wr3[kb + half * 16 + j];
    }
    acc0 = __builtin_amdgcn_wmma_f32_16x16x32_f16(false, a, false, b0, (short)0, acc0, false, false);
    acc1 = __builtin_amdgcn_wmma_f32_16x16x32_f16(false, a, false, b1, (short)0, acc1, false, false);
    acc2 = __builtin_amdgcn_wmma_f32_16x16x32_f16(false, a, false, b2, (short)0, acc2, false, false);
    acc3 = __builtin_amdgcn_wmma_f32_16x16x32_f16(false, a, false, b3, (short)0, acc3, false, false);
  }
  #pragma unroll
  for (int r = 0; r < 8; ++r) {
    int m = m0 + half * 8 + r;
    float rscale = rowscale ? rowscale[(size_t)m * rs_stride] : 1.f;
    float vals[4] = { acc0[r], acc1[r], acc2[r], acc3[r] };
    #pragma unroll
    for (int t = 0; t < 4; ++t) {
      int n = n0 + t * 16 + mr;
      float v = vals[t];
      if (bias)     v += bias[n];
      v *= rscale;
      size_t idx = (size_t)m * N + n;
      float o = v;
      if (residual)   o += residual[idx];
      if (accumulate) o += C[idx];
      C[idx] = o;
    }
  }
}

// ---------------- RoPE (in-place) ----------------
__global__ __launch_bounds__(256) void rope_kernel(
    float* __restrict__ buf, int nheads,
    const float* __restrict__ cosT, const float* __restrict__ sinT,
    const int* __restrict__ pos) {
  int idx = blockIdx.x * blockDim.x + threadIdx.x;   // over S * nheads * 32 pairs
  int total = S_LEN * nheads * (DHEAD / 2);
  if (idx >= total) return;
  int d = idx & 31;
  int h = (idx >> 5) % nheads;
  int s = idx / (32 * nheads);
  int p = pos[s];
  float c  = cosT[p * DHEAD + d];    // cos[p][d] == cos[p][d+32]
  float sn = sinT[p * DHEAD + d];
  float* base = buf + (size_t)s * (nheads * DHEAD) + h * DHEAD;
  float x0 = base[d], x1 = base[d + 32];
  base[d]      = x0 * c - x1 * sn;
  base[d + 32] = x1 * c + x0 * sn;
}

// ---------------- Flash attention (causal, GQA), WMMA QK^T and PV ----------------
__global__ __launch_bounds__(256) void attn_kernel(
    const float* __restrict__ Q, const float* __restrict__ Kb,
    const float* __restrict__ Vb, float* __restrict__ O) {
  __shared__ _Float16 plds[8][16][32];      // per-wave P tile (16 rows x 32 keys)
  int lane = threadIdx.x & 31;
  int wv   = threadIdx.x >> 5;
  int half = lane >> 4;
  int mr   = lane & 15;
  int h = blockIdx.y;
  int g = h >> 2;                           // kv head (32/8 = 4x repeat)
  int m0 = blockIdx.x * 128 + wv * 16;
  const float scale = 0.125f;               // 1/sqrt(64)

  // Q fragments for d=[0,32) and d=[32,64), pre-scaled by 1/sqrt(D)
  const float* qrow = Q + (size_t)(m0 + mr) * HDIM + h * DHEAD;
  v16h aq0, aq1;
  #pragma unroll
  for (int j = 0; j < 8; ++j) {
    aq0[j]     = (_Float16)(qrow[half * 16 + j] * scale);
    aq0[j + 8] = (_Float16)(qrow[half * 16 + 8 + j] * scale);
    aq1[j]     = (_Float16)(qrow[32 + half * 16 + j] * scale);
    aq1[j + 8] = (_Float16)(qrow[32 + half * 16 + 8 + j] * scale);
  }

  v8f o0 = {}, o1 = {}, o2 = {}, o3 = {};
  float mrow[8], lrow[8];
  #pragma unroll
  for (int r = 0; r < 8; ++r) { mrow[r] = -1e30f; lrow[r] = 0.f; }

  int qend = m0 + 15;
  for (int kb = 0; kb <= qend; kb += 32) {
    v16h b;
    v8f s1 = {}, s2 = {};
    const float* k1 = Kb + (size_t)(kb + mr) * KVD + g * DHEAD;   // key row = kb+mr
    #pragma unroll
    for (int j = 0; j < 16; ++j) b[j] = (_Float16)k1[half * 16 + j];
    s1 = __builtin_amdgcn_wmma_f32_16x16x32_f16(false, aq0, false, b, (short)0, s1, false, false);
    #pragma unroll
    for (int j = 0; j < 16; ++j) b[j] = (_Float16)k1[32 + half * 16 + j];
    s1 = __builtin_amdgcn_wmma_f32_16x16x32_f16(false, aq1, false, b, (short)0, s1, false, false);
    const float* k2 = k1 + (size_t)16 * KVD;
    #pragma unroll
    for (int j = 0; j < 16; ++j) b[j] = (_Float16)k2[half * 16 + j];
    s2 = __builtin_amdgcn_wmma_f32_16x16x32_f16(false, aq0, false, b, (short)0, s2, false, false);
    #pragma unroll
    for (int j = 0; j < 16; ++j) b[j] = (_Float16)k2[32 + half * 16 + j];
    s2 = __builtin_amdgcn_wmma_f32_16x16x32_f16(false, aq1, false, b, (short)0, s2, false, false);

    int key1 = kb + mr, key2 = kb + 16 + mr;
    float p1v[8], p2v[8];
    #pragma unroll
    for (int r = 0; r < 8; ++r) {
      int qpos = m0 + half * 8 + r;
      float v1 = (key1 <= qpos) ? s1[r] : -1e30f;
      float v2 = (key2 <= qpos) ? s2[r] : -1e30f;
      float tmax = fmaxf(v1, v2);
      #pragma unroll
      for (int off = 1; off < 16; off <<= 1) tmax = fmaxf(tmax, __shfl_xor(tmax, off, 32));
      float mnew  = fmaxf(mrow[r], tmax);
      float alpha = __expf(mrow[r] - mnew);
      float p1 = __expf(v1 - mnew);
      float p2 = __expf(v2 - mnew);
      float rsum = p1 + p2;
      #pragma unroll
      for (int off = 1; off < 16; off <<= 1) rsum += __shfl_xor(rsum, off, 32);
      lrow[r] = lrow[r] * alpha + rsum;
      mrow[r] = mnew;
      o0[r] *= alpha; o1[r] *= alpha; o2[r] *= alpha; o3[r] *= alpha;
      p1v[r] = p1; p2v[r] = p2;
    }

    // transpose P from C-layout into A-fragment layout via per-wave LDS tile
    #pragma unroll
    for (int r = 0; r < 8; ++r) {
      plds[wv][half * 8 + r][mr]      = (_Float16)p1v[r];
      plds[wv][half * 8 + r][16 + mr] = (_Float16)p2v[r];
    }
    v16h ap;   // same-wave DS ops are in-order; compiler inserts s_wait_dscnt
    #pragma unroll
    for (int j = 0; j < 8; ++j) {
      ap[j]     = plds[wv][mr][half * 16 + j];
      ap[j + 8] = plds[wv][mr][half * 16 + 8 + j];
    }

    // O += P @ V  (V fragment: B[k][n] = V[key=kb+half*16+j][dim=dt*16+mr])
    const float* vbase = Vb + (size_t)(kb + half * 16) * KVD + g * DHEAD;
    v16h bv;
    #pragma unroll
    for (int j = 0; j < 16; ++j) bv[j] = (_Float16)vbase[(size_t)j * KVD + mr];
    o0 = __builtin_amdgcn_wmma_f32_16x16x32_f16(false, ap, false, bv, (short)0, o0, false, false);
    #pragma unroll
    for (int j = 0; j < 16; ++j) bv[j] = (_Float16)vbase[(size_t)j * KVD + 16 + mr];
    o1 = __builtin_amdgcn_wmma_f32_16x16x32_f16(false, ap, false, bv, (short)0, o1, false, false);
    #pragma unroll
    for (int j = 0; j < 16; ++j) bv[j] = (_Float16)vbase[(size_t)j * KVD + 32 + mr];
    o2 = __builtin_amdgcn_wmma_f32_16x16x32_f16(false, ap, false, bv, (short)0, o2, false, false);
    #pragma unroll
    for (int j = 0; j < 16; ++j) bv[j] = (_Float16)vbase[(size_t)j * KVD + 48 + mr];
    o3 = __builtin_amdgcn_wmma_f32_16x16x32_f16(false, ap, false, bv, (short)0, o3, false, false);
  }

  #pragma unroll
  for (int r = 0; r < 8; ++r) {
    int m = m0 + half * 8 + r;
    float invl = 1.f / lrow[r];
    size_t base = (size_t)m * HDIM + h * DHEAD;
    O[base + mr]      = o0[r] * invl;
    O[base + 16 + mr] = o1[r] * invl;
    O[base + 32 + mr] = o2[r] * invl;
    O[base + 48 + mr] = o3[r] * invl;
  }
}

// ---------------- Router: logits[t,e] = dot(x[t], gate_w[e]) ----------------
__global__ __launch_bounds__(256) void router_kernel(
    const float* __restrict__ x, const float* __restrict__ gw,
    float* __restrict__ logits) {
  int t = blockIdx.x;
  int e = threadIdx.x >> 5;
  int lane = threadIdx.x & 31;
  const float* xr = x + (size_t)t * HDIM;
  const float* wr = gw + (size_t)e * HDIM;
  float s = 0.f;
  for (int i = lane; i < HDIM; i += 32) s += xr[i] * wr[i];
  s = warp_reduce_sum32(s);
  if (lane == 0) logits[t * NEXP + e] = s;
}

// ---------------- Sparsemixer (inference branch, top-2) ----------------
__global__ __launch_bounds__(256) void sparsemixer_kernel(
    const float* __restrict__ logits, float* __restrict__ coeff) {
  int t = blockIdx.x * blockDim.x + threadIdx.x;
  if (t >= S_LEN) return;
  float s[NEXP];
  #pragma unroll
  for (int e = 0; e < NEXP; ++e) s[e] = logits[t * NEXP + e];
  float max1 = s[0]; int sel1 = 0;
  #pragma unroll
  for (int e = 1; e < NEXP; ++e) if (s[e] > max1) { max1 = s[e]; sel1 = e; }
  float den1 = 0.f;
  #pragma unroll
  for (int e = 0; e < NEXP; ++e) {
    float factor = fmaxf(fabsf(s[e]), max1);
    bool mask = (max1 - s[e]) > 2.f * JITTER * factor;
    if (!mask) den1 += __expf(s[e] - max1);
  }
  float m1 = 1.f / den1;                         // exp(s[sel1]-max1)==1
  float max2 = -1e30f; int sel2 = (sel1 == 0) ? 1 : 0;
  #pragma unroll
  for (int e = 0; e < NEXP; ++e) if (e != sel1 && s[e] > max2) { max2 = s[e]; sel2 = e; }
  float den2 = 0.f;
  #pragma unroll
  for (int e = 0; e < NEXP; ++e) {
    if (e == sel1) continue;                     // masked to -inf -> contributes 0
    float factor = fmaxf(fabsf(s[e]), max2);
    bool mask = (max2 - s[e]) > 2.f * JITTER * factor;
    if (!mask) den2 += __expf(s[e] - max2);
  }
  float m2 = 1.f / den2;
  #pragma unroll
  for (int e = 0; e < NEXP; ++e)
    coeff[t * NEXP + e] = (e == sel1 ? m1 : 0.f) + (e == sel2 ? m2 : 0.f);
}

// ---------------- hmid = silu(h1) * h3 (in-place into h1 allowed) ----------------
__global__ __launch_bounds__(256) void silu_mul_kernel(
    const float* __restrict__ h1, const float* __restrict__ h3,
    float* __restrict__ out, int n) {
  int i = blockIdx.x * blockDim.x + threadIdx.x;
  if (i >= n) return;
  float a = h1[i];
  out[i] = a * (1.f / (1.f + __expf(-a))) * h3[i];
}

extern "C" void kernel_launch(void* const* d_in, const int* in_sizes, int n_in,
                              void* d_out, int out_size, void* d_ws, size_t ws_size,
                              hipStream_t stream) {
  const float* hidden = (const float*)d_in[0];
  const float* ln1w = (const float*)d_in[1];
  const float* ln1b = (const float*)d_in[2];
  const float* ln2w = (const float*)d_in[3];
  const float* ln2b = (const float*)d_in[4];
  const float* qw = (const float*)d_in[5];
  const float* qb = (const float*)d_in[6];
  const float* kw = (const float*)d_in[7];
  const float* kb = (const float*)d_in[8];
  const float* vw = (const float*)d_in[9];
  const float* vb = (const float*)d_in[10];
  const float* ow = (const float*)d_in[11];
  const float* ob = (const float*)d_in[12];
  const float* gw = (const float*)d_in[13];
  const float* w1 = (const float*)d_in[14];
  const float* w3 = (const float*)d_in[15];
  const float* w2 = (const float*)d_in[16];
  const float* cosT = (const float*)d_in[17];
  const float* sinT = (const float*)d_in[18];
  const int* pos = (const int*)d_in[19];
  float* out = (float*)d_out;

  float* ws = (float*)d_ws;
  size_t off = 0;
  float* xln   = ws + off; off += (size_t)S_LEN * HDIM;
  float* qbuf  = ws + off; off += (size_t)S_LEN * HDIM;
  float* kbuf  = ws + off; off += (size_t)S_LEN * KVD;
  float* vbuf  = ws + off; off += (size_t)S_LEN * KVD;
  float* attnb = ws + off; off += (size_t)S_LEN * HDIM;
  float* hid1  = ws + off; off += (size_t)S_LEN * HDIM;
  float* xln2  = ws + off; off += (size_t)S_LEN * HDIM;
  float* logit = ws + off; off += (size_t)S_LEN * NEXP;
  float* coeff = ws + off; off += (size_t)S_LEN * NEXP;
  float* h1b   = ws + off; off += (size_t)S_LEN * IDIM;   // also holds hmid
  float* h3b   = ws + off; off += (size_t)S_LEN * IDIM;

  dim3 blk(256);

  // pre-attn LN
  ln_kernel<<<S_LEN, blk, 0, stream>>>(hidden, ln1w, ln1b, xln);
  // Q/K/V projections
  gemm_wmma_f16<<<dim3(HDIM / 512, S_LEN / 16), blk, 0, stream>>>(xln, qw, qb, nullptr, nullptr, 0, qbuf, S_LEN, HDIM, HDIM, 0);
  gemm_wmma_f16<<<dim3(KVD  / 512, S_LEN / 16), blk, 0, stream>>>(xln, kw, kb, nullptr, nullptr, 0, kbuf, S_LEN, KVD,  HDIM, 0);
  gemm_wmma_f16<<<dim3(KVD  / 512, S_LEN / 16), blk, 0, stream>>>(xln, vw, vb, nullptr, nullptr, 0, vbuf, S_LEN, KVD,  HDIM, 0);
  // RoPE
  rope_kernel<<<(S_LEN * NHEADS  * 32 + 255) / 256, blk, 0, stream>>>(qbuf, NHEADS,  cosT, sinT, pos);
  rope_kernel<<<(S_LEN * KVHEADS * 32 + 255) / 256, blk, 0, stream>>>(kbuf, KVHEADS, cosT, sinT, pos);
  // causal GQA flash attention
  attn_kernel<<<dim3(S_LEN / 128, NHEADS), blk, 0, stream>>>(qbuf, kbuf, vbuf, attnb);
  // O projection + residual
  gemm_wmma_f16<<<dim3(HDIM / 512, S_LEN / 16), blk, 0, stream>>>(attnb, ow, ob, hidden, nullptr, 0, hid1, S_LEN, HDIM, HDIM, 0);
  // post-attn LN
  ln_kernel<<<S_LEN, blk, 0, stream>>>(hid1, ln2w, ln2b, xln2);
  // router + sparsemixer
  router_kernel<<<S_LEN, blk, 0, stream>>>(xln2, gw, logit);
  sparsemixer_kernel<<<(S_LEN + 255) / 256, blk, 0, stream>>>(logit, coeff);
  // MoE: dense over experts, masked by coeff (zero coeff => zero contribution)
  for (int e = 0; e < NEXP; ++e) {
    const float* w1e = w1 + (size_t)e * IDIM * HDIM;
    const float* w3e = w3 + (size_t)e * IDIM * HDIM;
    const float* w2e = w2 + (size_t)e * HDIM * IDIM;
    gemm_wmma_f16<<<dim3(IDIM / 512, S_LEN / 16), blk, 0, stream>>>(xln2, w1e, nullptr, nullptr, nullptr, 0, h1b, S_LEN, IDIM, HDIM, 0);
    gemm_wmma_f16<<<dim3(IDIM / 512, S_LEN / 16), blk, 0, stream>>>(xln2, w3e, nullptr, nullptr, nullptr, 0, h3b, S_LEN, IDIM, HDIM, 0);
    silu_mul_kernel<<<((S_LEN * IDIM) + 255) / 256, blk, 0, stream>>>(h1b, h3b, h1b, S_LEN * IDIM);
    // out = hid1 + sum_e coeff[:,e] * (hmid @ w2[e]^T)
    gemm_wmma_f16<<<dim3(HDIM / 512, S_LEN / 16), blk, 0, stream>>>(h1b, w2e, nullptr,
        (e == 0) ? hid1 : nullptr, coeff + e, NEXP, out, S_LEN, HDIM, IDIM, (e == 0) ? 0 : 1);
  }
}
